// BasicTransformerBlockST_51410758533171
// MI455X (gfx1250) — compile-verified, matrix operand order checked by reading, not run on hardware
//
#include <hip/hip_runtime.h>
#include <hip/hip_bf16.h>
#include <math.h>

typedef __attribute__((ext_vector_type(16))) _Float16 v16h;
typedef __attribute__((ext_vector_type(8)))  _Float16 v8h;
typedef __attribute__((ext_vector_type(8)))  float    v8f;
typedef __attribute__((ext_vector_type(4)))  float    v4f;

// Problem constants
#define DD   320
#define CTXD 768
#define NHEADS 8
#define DH   40
#define TLEN 16
#define HH   32
#define WW2  32
#define NB   2
#define NTOK (NB*TLEN*HH*WW2)      // 32768
#define FFD  1280
#define MAXR 16

// ---------------------------------------------------------------------------
// WMMA GEMM: C[M,N] = A[M,K] @ W[K,N] (+bias[N]) (+resid[M,N])
// fp32 in/out, f16 tiles, fp32 accumulate via v_wmma_f32_16x16x32_f16.
// Block tile 128x64, BK=32; 256 threads = 8 waves (4x2), each wave computes a
// 32x32 register tile: 2 A-frags x 2 B-frags -> 4 WMMAs per K-step
// (8 ds_load_b128 per 4 WMMAs). B tile stored transposed so every lane's
// fragment halves are two contiguous 16B runs.
// ---------------------------------------------------------------------------
#define BM 128
#define BN 64
#define BK 32

__global__ __launch_bounds__(256) void gemm_wmma_k(
    const float* __restrict__ A, const float* __restrict__ W,
    const float* __restrict__ bias, const float* __restrict__ resid,
    float* __restrict__ C, int M, int K, int N)
{
  __shared__ alignas(16) _Float16 As [BM][BK];   // [m][k]  row = 64B
  __shared__ alignas(16) _Float16 BsT[BN][BK];   // [n][k]  row = 64B
  const int tid  = threadIdx.x;
  const int lane = tid & 31;
  const int wave = tid >> 5;
  const int wm   = wave >> 1;          // 0..3  (32-row block)
  const int wn   = wave & 1;           // 0..1  (32-col block)
  const int m0   = blockIdx.x * BM;
  const int n0   = blockIdx.y * BN;

  const int half = lane >> 4;          // 0/1
  const int l16  = lane & 15;

  // staging assignments (branch-free, vectorized)
  const int ra  = tid >> 1;            // 0..127 : A row
  const int ca  = (tid & 1) * 16;      // 0/16   : A col start (16 floats)
  const int rb  = tid >> 3;            // 0..31  : B row (k)
  const int cb  = (tid & 7) * 8;       // 0..56  : B col start (8 floats)

  v8f acc00 = {}, acc01 = {}, acc10 = {}, acc11 = {};

  const int mrow0 = wm * 32 + l16;
  const int mrow1 = mrow0 + 16;
  const int nc0   = wn * 32 + l16;
  const int nc1   = nc0 + 16;

  for (int k0 = 0; k0 < K; k0 += BK) {
    // ---- stage A tile: 128x32 fp32 -> f16 (16 floats/thread) ----
    {
      int gm = m0 + ra; gm = gm < M ? gm : (M - 1);   // clamp, no branch
      const v4f* ap = (const v4f*)(A + (size_t)gm * K + k0 + ca);
      v4f a0 = ap[0], a1 = ap[1], a2 = ap[2], a3 = ap[3];
      v8h h0, h1;
      #pragma unroll
      for (int i = 0; i < 4; ++i) {
        h0[i] = (_Float16)a0[i]; h0[4+i] = (_Float16)a1[i];
        h1[i] = (_Float16)a2[i]; h1[4+i] = (_Float16)a3[i];
      }
      *(v8h*)&As[ra][ca]     = h0;
      *(v8h*)&As[ra][ca + 8] = h1;
    }
    // ---- stage B tile transposed: BsT[n][k] = W[k0+k][n0+n] ----
    {
      const v4f* bp = (const v4f*)(W + (size_t)(k0 + rb) * N + n0 + cb);
      v4f b0 = bp[0], b1 = bp[1];
      #pragma unroll
      for (int i = 0; i < 4; ++i) {
        BsT[cb + i    ][rb] = (_Float16)b0[i];
        BsT[cb + 4 + i][rb] = (_Float16)b1[i];
      }
    }
    if (k0 + BK < K) __builtin_prefetch(&W[(size_t)(k0 + BK) * N + n0], 0, 1);
    __syncthreads();

    // ---- fragments: two contiguous 16B runs per lane -> ds_load_b128 ----
    const v8h* ar0 = (const v8h*)&As[mrow0][0];
    v8h a00 = ar0[half], a01 = ar0[2 + half];
    v16h af0 = __builtin_shufflevector(a00, a01, 0,1,2,3,4,5,6,7,8,9,10,11,12,13,14,15);

    const v8h* ar1 = (const v8h*)&As[mrow1][0];
    v8h a10 = ar1[half], a11 = ar1[2 + half];
    v16h af1 = __builtin_shufflevector(a10, a11, 0,1,2,3,4,5,6,7,8,9,10,11,12,13,14,15);

    const v8h* br0 = (const v8h*)&BsT[nc0][0];
    v8h b00 = br0[half], b01 = br0[2 + half];
    v16h bf0 = __builtin_shufflevector(b00, b01, 0,1,2,3,4,5,6,7,8,9,10,11,12,13,14,15);

    const v8h* br1 = (const v8h*)&BsT[nc1][0];
    v8h b10 = br1[half], b11 = br1[2 + half];
    v16h bf1 = __builtin_shufflevector(b10, b11, 0,1,2,3,4,5,6,7,8,9,10,11,12,13,14,15);

    acc00 = __builtin_amdgcn_wmma_f32_16x16x32_f16(false, af0, false, bf0, (short)0, acc00, false, false);
    acc01 = __builtin_amdgcn_wmma_f32_16x16x32_f16(false, af0, false, bf1, (short)0, acc01, false, false);
    acc10 = __builtin_amdgcn_wmma_f32_16x16x32_f16(false, af1, false, bf0, (short)0, acc10, false, false);
    acc11 = __builtin_amdgcn_wmma_f32_16x16x32_f16(false, af1, false, bf1, (short)0, acc11, false, false);
    __syncthreads();
  }

  // ---- epilogue: D layout -> M = v + 8*half, N = l16 ----
  const int gn0 = n0 + wn * 32 + l16;
  const int gn1 = gn0 + 16;
  const float bv0 = bias ? bias[gn0] : 0.f;
  const float bv1 = bias ? bias[gn1] : 0.f;
  #pragma unroll
  for (int v = 0; v < 8; ++v) {
    int gm = m0 + wm * 32 + v + 8 * half;
    if (gm < M) {
      size_t i0 = (size_t)gm * N + gn0;
      size_t i1 = (size_t)gm * N + gn1;
      float v0 = acc00[v] + bv0;
      float v1 = acc01[v] + bv1;
      if (resid) { v0 += resid[i0]; v1 += resid[i1]; }
      C[i0] = v0; C[i1] = v1;
    }
    int gm2 = gm + 16;
    if (gm2 < M) {
      size_t i0 = (size_t)gm2 * N + gn0;
      size_t i1 = (size_t)gm2 * N + gn1;
      float v0 = acc10[v] + bv0;
      float v1 = acc11[v] + bv1;
      if (resid) { v0 += resid[i0]; v1 += resid[i1]; }
      C[i0] = v0; C[i1] = v1;
    }
  }
}

// ---------------------------------------------------------------------------
// LayerNorm: one wave32 per token of 320 elements (10/lane).
// ---------------------------------------------------------------------------
__global__ __launch_bounds__(256) void layernorm_k(
    const float* __restrict__ X, const float* __restrict__ g,
    const float* __restrict__ b, float* __restrict__ Y, int rows)
{
  int wave = threadIdx.x >> 5, lane = threadIdx.x & 31;
  int row = blockIdx.x * 8 + wave;
  if (row >= rows) return;
  const float* xr = X + (size_t)row * DD;
  float vals[10];
  float s = 0.f;
  #pragma unroll
  for (int i = 0; i < 10; ++i) { vals[i] = xr[lane + 32*i]; s += vals[i]; }
  #pragma unroll
  for (int off = 16; off > 0; off >>= 1) s += __shfl_xor(s, off, 32);
  float mu = s * (1.f / DD);
  float v2 = 0.f;
  #pragma unroll
  for (int i = 0; i < 10; ++i) { float d = vals[i] - mu; v2 += d * d; }
  #pragma unroll
  for (int off = 16; off > 0; off >>= 1) v2 += __shfl_xor(v2, off, 32);
  float rstd = rsqrtf(v2 * (1.f / DD) + 1e-5f);
  float* yr = Y + (size_t)row * DD;
  #pragma unroll
  for (int i = 0; i < 10; ++i) {
    int c = lane + 32*i;
    yr[c] = (vals[i] - mu) * rstd * g[c] + b[c];
  }
}

// ---------------------------------------------------------------------------
// Windowed self-attention. grid = (128 windows, 8 heads), 256 threads = 256
// queries per window. K/V staged in LDS in 2 tiles of 128 keys, online softmax.
// ---------------------------------------------------------------------------
__global__ __launch_bounds__(256) void attn_win_k(
    const float* __restrict__ Q, const float* __restrict__ K,
    const float* __restrict__ V, float* __restrict__ O)
{
  __shared__ float sk[128][DH];
  __shared__ float sv[128][DH];
  const int win  = blockIdx.x;
  const int head = blockIdx.y;
  const int b    = win >> 6;
  const int wrem = win & 63;
  const int ih   = wrem >> 3, iw = wrem & 7;
  const int tid  = threadIdx.x;
  const int tq   = tid >> 4;
  const int r    = tid & 15;
  const int hq   = ih * 4 + (r >> 2), wq = iw * 4 + (r & 3);
  const int qtok = ((b * TLEN + tq) * HH + hq) * WW2 + wq;
  const int coff = head * DH;
  const float scale = 0.1581138830084190f;   // 1/sqrt(40)

  float q[DH], acc[DH];
  #pragma unroll
  for (int d = 0; d < DH; ++d) { q[d] = Q[(size_t)qtok * DD + coff + d]; acc[d] = 0.f; }
  float m = -1e30f, l = 0.f;

  for (int tile = 0; tile < 2; ++tile) {
    __syncthreads();
    for (int e = tid; e < 128 * DH; e += 256) {
      int j = e / DH, d = e - j * DH;
      int jj = tile * 128 + j;
      int tk = jj >> 4, rr = jj & 15;
      int kh = ih * 4 + (rr >> 2), kw = iw * 4 + (rr & 3);
      int ktok = ((b * TLEN + tk) * HH + kh) * WW2 + kw;
      sk[j][d] = K[(size_t)ktok * DD + coff + d];
      sv[j][d] = V[(size_t)ktok * DD + coff + d];
    }
    __syncthreads();
    for (int j = 0; j < 128; ++j) {
      float s = 0.f;
      #pragma unroll
      for (int d = 0; d < DH; ++d) s += q[d] * sk[j][d];
      s *= scale;
      float nm = fmaxf(m, s);
      float corr = __expf(m - nm);
      float p = __expf(s - nm);
      l = l * corr + p;
      #pragma unroll
      for (int d = 0; d < DH; ++d) acc[d] = acc[d] * corr + p * sv[j][d];
      m = nm;
    }
  }
  float inv = 1.f / l;
  #pragma unroll
  for (int d = 0; d < DH; ++d) O[(size_t)qtok * DD + coff + d] = acc[d] * inv;
}

// ---------------------------------------------------------------------------
// Temporal self-attn with rel-pos bias + causal mask. grid = (128 seq-groups,
// 8 heads), 256 threads = 16 sequences x 16 query times.
// ---------------------------------------------------------------------------
__global__ __launch_bounds__(256) void attn_tmp_k(
    const float* __restrict__ Q, const float* __restrict__ K,
    const float* __restrict__ V, const float* __restrict__ relk,
    const float* __restrict__ relv, float* __restrict__ O)
{
  const int head = blockIdx.y;
  const int tid  = threadIdx.x;
  const int seq  = blockIdx.x * 16 + (tid >> 4);
  const int i    = tid & 15;
  const int b    = seq >> 10;
  const int hw   = seq & 1023;
  const int base = ((b * TLEN) * HH + (hw >> 5)) * WW2 + (hw & 31);
  const int coff = head * DH;
  const float scale = 0.1581138830084190f;

  const int qtok = base + i * (HH * WW2);
  float q[DH], acc[DH];
  #pragma unroll
  for (int d = 0; d < DH; ++d) { q[d] = Q[(size_t)qtok * DD + coff + d]; acc[d] = 0.f; }
  float m = -1e30f, l = 0.f;

  for (int j = 0; j <= i; ++j) {
    int ktok = base + j * (HH * WW2);
    const float* kr = K + (size_t)ktok * DD + coff;
    const float* vr = V + (size_t)ktok * DD + coff;
    const float* rk = relk + (size_t)(j - i + MAXR) * DH;
    const float* rv = relv + (size_t)(j - i + MAXR) * DH;
    float s = 0.f;
    #pragma unroll
    for (int d = 0; d < DH; ++d) s += q[d] * (kr[d] + rk[d]);
    s *= scale;
    float nm = fmaxf(m, s);
    float corr = __expf(m - nm);
    float p = __expf(s - nm);
    l = l * corr + p;
    #pragma unroll
    for (int d = 0; d < DH; ++d) acc[d] = acc[d] * corr + p * (vr[d] + rv[d]);
    m = nm;
  }
  float inv = 1.f / l;
  #pragma unroll
  for (int d = 0; d < DH; ++d) O[(size_t)qtok * DD + coff + d] = acc[d] * inv;
}

// ---------------------------------------------------------------------------
// Cross-attn to text context. grid = (32 (b,t) * 4 query-chunks, 8 heads),
// 256 threads = 256 queries; 77 keys staged in LDS.
// ---------------------------------------------------------------------------
__global__ __launch_bounds__(256) void attn_cross_k(
    const float* __restrict__ Q, const float* __restrict__ KC,
    const float* __restrict__ VC, float* __restrict__ O)
{
  __shared__ float sk[77][DH];
  __shared__ float sv[77][DH];
  const int head = blockIdx.y;
  const int bt   = blockIdx.x >> 2;
  const int qc   = blockIdx.x & 3;
  const int b    = bt >> 4;
  const int tid  = threadIdx.x;
  const int coff = head * DH;
  const float scale = 0.1581138830084190f;

  for (int e = tid; e < 77 * DH; e += 256) {
    int j = e / DH, d = e - j * DH;
    sk[j][d] = KC[(size_t)(b * 77 + j) * DD + coff + d];
    sv[j][d] = VC[(size_t)(b * 77 + j) * DD + coff + d];
  }
  __syncthreads();

  const int qrow = bt * 1024 + qc * 256 + tid;
  float q[DH], acc[DH];
  #pragma unroll
  for (int d = 0; d < DH; ++d) { q[d] = Q[(size_t)qrow * DD + coff + d]; acc[d] = 0.f; }
  float m = -1e30f, l = 0.f;
  for (int j = 0; j < 77; ++j) {
    float s = 0.f;
    #pragma unroll
    for (int d = 0; d < DH; ++d) s += q[d] * sk[j][d];
    s *= scale;
    float nm = fmaxf(m, s);
    float corr = __expf(m - nm);
    float p = __expf(s - nm);
    l = l * corr + p;
    #pragma unroll
    for (int d = 0; d < DH; ++d) acc[d] = acc[d] * corr + p * sv[j][d];
    m = nm;
  }
  float inv = 1.f / l;
  #pragma unroll
  for (int d = 0; d < DH; ++d) O[(size_t)qrow * DD + coff + d] = acc[d] * inv;
}

// ---------------------------------------------------------------------------
// GEGLU gate: G[r,c] = H[r,c] * gelu_exact(H[r,1280+c]); H is [Mc, 2560]
// ---------------------------------------------------------------------------
__global__ __launch_bounds__(256) void geglu_k(
    const float* __restrict__ Hh, float* __restrict__ G, int total)
{
  int i = blockIdx.x * 256 + threadIdx.x;
  if (i >= total) return;
  int row = i / FFD, c = i - row * FFD;
  float a = Hh[(size_t)row * (2 * FFD) + c];
  float g = Hh[(size_t)row * (2 * FFD) + FFD + c];
  float ge = 0.5f * g * (1.f + erff(g * 0.70710678118654752f));
  G[i] = a * ge;
}

// ---------------------------------------------------------------------------
// Layout transposes: [B,C,T,H,W] <-> token-major [B,T,H,W,C]
// ---------------------------------------------------------------------------
__global__ __launch_bounds__(256) void nchw_to_tok_k(
    const float* __restrict__ X, float* __restrict__ Y, int total)
{
  int i = blockIdx.x * 256 + threadIdx.x;
  if (i >= total) return;
  int c = i % DD; int t2 = i / DD;
  int w = t2 % WW2; t2 /= WW2;
  int h = t2 % HH;  t2 /= HH;
  int t = t2 % TLEN; int b = t2 / TLEN;
  Y[i] = X[((((size_t)b * DD + c) * TLEN + t) * HH + h) * WW2 + w];
}

__global__ __launch_bounds__(256) void tok_to_nchw_k(
    const float* __restrict__ X, float* __restrict__ Y, int total)
{
  int i = blockIdx.x * 256 + threadIdx.x;
  if (i >= total) return;
  int w = i % WW2; int t2 = i / WW2;
  int h = t2 % HH;  t2 /= HH;
  int t = t2 % TLEN; t2 /= TLEN;
  int c = t2 % DD;  int b = t2 / DD;
  Y[i] = X[((((size_t)b * TLEN + t) * HH + h) * WW2 + w) * DD + c];
}

// ---------------------------------------------------------------------------
// Host orchestration
// ---------------------------------------------------------------------------
static inline void gemm(const float* A, const float* W, const float* bias,
                        const float* resid, float* C, int M, int K, int N,
                        hipStream_t s)
{
  dim3 grid((M + BM - 1) / BM, N / BN);
  gemm_wmma_k<<<grid, 256, 0, s>>>(A, W, bias, resid, C, M, K, N);
}

extern "C" void kernel_launch(void* const* d_in, const int* in_sizes, int n_in,
                              void* d_out, int out_size, void* d_ws, size_t ws_size,
                              hipStream_t stream)
{
  const float* x    = (const float*)d_in[0];
  const float* ctx  = (const float*)d_in[1];
  const float* ln1g = (const float*)d_in[2];  const float* ln1b = (const float*)d_in[3];
  const float* ln2g = (const float*)d_in[4];  const float* ln2b = (const float*)d_in[5];
  const float* ln3g = (const float*)d_in[6];  const float* ln3b = (const float*)d_in[7];
  const float* ln4g = (const float*)d_in[8];  const float* ln4b = (const float*)d_in[9];
  const float* ln5g = (const float*)d_in[10]; const float* ln5b = (const float*)d_in[11];
  const float* a1Wq = (const float*)d_in[12]; const float* a1Wk = (const float*)d_in[13];
  const float* a1Wv = (const float*)d_in[14]; const float* a1Wo = (const float*)d_in[15];
  const float* a1bo = (const float*)d_in[16];
  const float* a2Wq = (const float*)d_in[17]; const float* a2Wk = (const float*)d_in[18];
  const float* a2Wv = (const float*)d_in[19]; const float* a2Wo = (const float*)d_in[20];
  const float* a2bo = (const float*)d_in[21];
  const float* t1Wq = (const float*)d_in[22]; const float* t1Wk = (const float*)d_in[23];
  const float* t1Wv = (const float*)d_in[24]; const float* t1Wo = (const float*)d_in[25];
  const float* t1bo = (const float*)d_in[26];
  const float* t1rk = (const float*)d_in[27]; const float* t1rv = (const float*)d_in[28];
  const float* t2Wq = (const float*)d_in[29]; const float* t2Wk = (const float*)d_in[30];
  const float* t2Wv = (const float*)d_in[31]; const float* t2Wo = (const float*)d_in[32];
  const float* t2bo = (const float*)d_in[33];
  const float* t2rk = (const float*)d_in[34]; const float* t2rv = (const float*)d_in[35];
  const float* ffW1 = (const float*)d_in[36]; const float* ffb1 = (const float*)d_in[37];
  const float* ffW2 = (const float*)d_in[38]; const float* ffb2 = (const float*)d_in[39];

  const size_t NTD = (size_t)NTOK * DD;      // 10,485,760 floats
  float* ws = (float*)d_ws;
  float* xt = ws;                            // token-major activations
  float* nb = xt + NTD;                      // layernorm output
  float* qb = nb + NTD;
  float* kb = qb + NTD;
  float* vb = kb + NTD;
  float* ob = vb + NTD;
  float* kc = ob + NTD;                      // 154*320
  float* vc = kc + (size_t)154 * DD;
  float* hb = vc + (size_t)154 * DD;         // FF chunk: 4096*2560
  float* gb = hb + (size_t)4096 * 2 * FFD;   // FF gate chunk: 4096*1280

  const int total = (int)NTD;
  const int eltBlocks = (total + 255) / 256;
  const int lnBlocks  = (NTOK + 7) / 8;

  nchw_to_tok_k<<<eltBlocks, 256, 0, stream>>>(x, xt, total);

  // ---- stage 1: windowed self-attention -------------------------------
  layernorm_k<<<lnBlocks, 256, 0, stream>>>(xt, ln1g, ln1b, nb, NTOK);
  gemm(nb, a1Wq, nullptr, nullptr, qb, NTOK, DD, DD, stream);
  gemm(nb, a1Wk, nullptr, nullptr, kb, NTOK, DD, DD, stream);
  gemm(nb, a1Wv, nullptr, nullptr, vb, NTOK, DD, DD, stream);
  attn_win_k<<<dim3(128, NHEADS), 256, 0, stream>>>(qb, kb, vb, ob);
  gemm(ob, a1Wo, a1bo, xt, xt, NTOK, DD, DD, stream);

  // ---- stage 2: temporal self-attn #1 (causal + rel-pos) --------------
  layernorm_k<<<lnBlocks, 256, 0, stream>>>(xt, ln4g, ln4b, nb, NTOK);
  gemm(nb, t1Wq, nullptr, nullptr, qb, NTOK, DD, DD, stream);
  gemm(nb, t1Wk, nullptr, nullptr, kb, NTOK, DD, DD, stream);
  gemm(nb, t1Wv, nullptr, nullptr, vb, NTOK, DD, DD, stream);
  attn_tmp_k<<<dim3(128, NHEADS), 256, 0, stream>>>(qb, kb, vb, t1rk, t1rv, ob);
  gemm(ob, t1Wo, t1bo, xt, xt, NTOK, DD, DD, stream);

  // ---- stage 3: spatial cross-attn to context -------------------------
  layernorm_k<<<lnBlocks, 256, 0, stream>>>(xt, ln2g, ln2b, nb, NTOK);
  gemm(nb, a2Wq, nullptr, nullptr, qb, NTOK, DD, DD, stream);
  gemm(ctx, a2Wk, nullptr, nullptr, kc, NB * 77, CTXD, DD, stream);
  gemm(ctx, a2Wv, nullptr, nullptr, vc, NB * 77, CTXD, DD, stream);
  attn_cross_k<<<dim3(128, NHEADS), 256, 0, stream>>>(qb, kc, vc, ob);
  gemm(ob, a2Wo, a2bo, xt, xt, NTOK, DD, DD, stream);

  // ---- stage 4: temporal self-attn #2 ---------------------------------
  layernorm_k<<<lnBlocks, 256, 0, stream>>>(xt, ln5g, ln5b, nb, NTOK);
  gemm(nb, t2Wq, nullptr, nullptr, qb, NTOK, DD, DD, stream);
  gemm(nb, t2Wk, nullptr, nullptr, kb, NTOK, DD, DD, stream);
  gemm(nb, t2Wv, nullptr, nullptr, vb, NTOK, DD, DD, stream);
  attn_tmp_k<<<dim3(128, NHEADS), 256, 0, stream>>>(qb, kb, vb, t2rk, t2rv, ob);
  gemm(ob, t2Wo, t2bo, xt, xt, NTOK, DD, DD, stream);

  // ---- stage 5: GEGLU feed-forward (chunked over rows) ----------------
  layernorm_k<<<lnBlocks, 256, 0, stream>>>(xt, ln3g, ln3b, nb, NTOK);
  const int CH = 4096;
  for (int c0 = 0; c0 < NTOK; c0 += CH) {
    const float* Ach = nb + (size_t)c0 * DD;
    gemm(Ach, ffW1, ffb1, nullptr, hb, CH, DD, 2 * FFD, stream);
    int gtotal = CH * FFD;
    geglu_k<<<(gtotal + 255) / 256, 256, 0, stream>>>(hb, gb, gtotal);
    float* xch = xt + (size_t)c0 * DD;
    gemm(gb, ffW2, ffb2, xch, xch, CH, FFD, DD, stream);
  }

  tok_to_nchw_k<<<eltBlocks, 256, 0, stream>>>(xt, (float*)d_out, total);
}
